// _Interactions_23021024707091
// MI455X (gfx1250) — matrix operand order repaired and checked by reading, not compiled, not run on hardware
//
#include <hip/hip_runtime.h>
#include <hip/hip_bf16.h>

typedef __attribute__((ext_vector_type(16))) _Float16 v16h;
typedef __attribute__((ext_vector_type(8)))  float    v8f;

#define ALPHA 0.9f
#define FDIM 64

// ---------------------------------------------------------------------------
// degree scatter:  deg[col[e]] += w[e]
// ---------------------------------------------------------------------------
__global__ void deg_kernel(const long long* __restrict__ idx,
                           const float* __restrict__ ew,
                           float* __restrict__ deg, long long E) {
  long long e = (long long)blockIdx.x * blockDim.x + threadIdx.x;
  if (e >= E) return;
  long long c = idx[E + e];
  atomicAdd(&deg[c], ew[e]);
}

// ---------------------------------------------------------------------------
// norm_w[e] = dinv[row]*w*dinv[col],  dinv = deg>0 ? rsqrt(deg) : 0
// ---------------------------------------------------------------------------
__global__ void normw_kernel(const long long* __restrict__ idx,
                             const float* __restrict__ ew,
                             const float* __restrict__ deg,
                             float* __restrict__ nw, long long E) {
  long long e = (long long)blockIdx.x * blockDim.x + threadIdx.x;
  if (e >= E) return;
  long long r = idx[e], c = idx[E + e];
  float dr = deg[r], dc = deg[c];
  float ir = dr > 0.f ? rsqrtf(dr) : 0.f;
  float ic = dc > 0.f ? rsqrtf(dc) : 0.f;
  nw[e] = ir * ew[e] * ic;
}

// ---------------------------------------------------------------------------
// edge scatter: 16 threads per edge, each moves 4 floats.
//   agg[col] += h[row] * norm_w[e]
// h/agg (25.6 MB each) live in the 192 MB L2, so the random traffic stays
// on-chip; only idx/norm_w stream from HBM.
// ---------------------------------------------------------------------------
__global__ void scatter_kernel(const float* __restrict__ h,
                               const float* __restrict__ nw,
                               const long long* __restrict__ idx,
                               float* __restrict__ agg, long long E) {
  long long t = (long long)blockIdx.x * blockDim.x + threadIdx.x;
  long long e = t >> 4;
  int c = (int)(t & 15);
  if (e >= E) return;
  // stream-ahead prefetch of the edge-index arrays (global_prefetch_b8)
  __builtin_prefetch(idx + e + 4096, 0, 0);
  __builtin_prefetch(idx + E + e + 4096, 0, 0);
  long long r = idx[e];
  long long col = idx[E + e];
  float w = nw[e];
  const float4 v = *(const float4*)(h + (size_t)r * FDIM + c * 4);
  float* dst = agg + (size_t)col * FDIM + c * 4;
  atomicAdd(dst + 0, v.x * w);
  atomicAdd(dst + 1, v.y * w);
  atomicAdd(dst + 2, v.z * w);
  atomicAdd(dst + 3, v.w * w);
}

// ---------------------------------------------------------------------------
// t = (1-ALPHA)*agg + ALPHA*h0, in place on agg (float4 vectorized)
// ---------------------------------------------------------------------------
__global__ void axpby_kernel(float* __restrict__ agg,
                             const float* __restrict__ h0, size_t n4) {
  size_t i = (size_t)blockIdx.x * blockDim.x + threadIdx.x;
  if (i >= n4) return;
  float4 a = ((const float4*)agg)[i];
  float4 b = ((const float4*)h0)[i];
  float4 o;
  o.x = (1.f - ALPHA) * a.x + ALPHA * b.x;
  o.y = (1.f - ALPHA) * a.y + ALPHA * b.y;
  o.z = (1.f - ALPHA) * a.z + ALPHA * b.z;
  o.w = (1.f - ALPHA) * a.w + ALPHA * b.w;
  ((float4*)agg)[i] = o;
}

// ---------------------------------------------------------------------------
// WMMA GEMM:  out[m][n] = f(A[m][:] @ W[:,n])
//   if residual:  out = residual + relu(acc)
//   else:         out = relu(acc + bias[n])
// A: [nrows,64] f32.  W staged in LDS as f16, n-major ([n][k]) so each lane's
// B fragment is one contiguous 32-byte v16h read.
//   w_nk = 1: global W already [n][k]   (lin0_w is [F_out, F_in])
//   w_nk = 0: global W is [k][n]        (w1)
// One wave computes one 16x64 tile: K=64 -> 2 k-steps of 32, 4 n-tiles of 16
// -> 8 x v_wmma_f32_16x16x32_f16 per wave.
// ---------------------------------------------------------------------------
__device__ inline void load8h(v16h& a, int base, const float* p) {
  float4 u = *(const float4*)p;
  float4 v = *(const float4*)(p + 4);
  a[base + 0] = (_Float16)u.x; a[base + 1] = (_Float16)u.y;
  a[base + 2] = (_Float16)u.z; a[base + 3] = (_Float16)u.w;
  a[base + 4] = (_Float16)v.x; a[base + 5] = (_Float16)v.y;
  a[base + 6] = (_Float16)v.z; a[base + 7] = (_Float16)v.w;
}

__global__ void __launch_bounds__(256)
gemm_kernel(const float* __restrict__ A, const float* __restrict__ Wg,
            const float* __restrict__ bias, const float* __restrict__ residual,
            float* __restrict__ out, int nrows, int w_nk) {
  __shared__ _Float16 wt[FDIM * FDIM];  // [n][k], 8 KB

  // cooperative transpose/convert of the 64x64 weight into LDS
  for (int i = threadIdx.x; i < FDIM * FDIM; i += blockDim.x) {
    int n = i >> 6, k = i & 63;
    float v = w_nk ? Wg[n * FDIM + k] : Wg[k * FDIM + n];
    wt[n * FDIM + k] = (_Float16)v;
  }
  __syncthreads();

  const int wave = threadIdx.x >> 5;
  const int lane = threadIdx.x & 31;
  const int ntiles = (nrows + 15) >> 4;
  const int tile = blockIdx.x * 8 + wave;
  if (tile >= ntiles) return;

  const int row0 = tile * 16;
  // ---- A fragments (contiguous k order, 16x32 per k-step) ----
  //  lanes 0-15 : M=lane,    K = 0..15 (+kstep)
  //  lanes 16-31: M=lane-16, K = 16..31 (+kstep)
  int m = lane & 15;
  int mrow = row0 + m;
  if (mrow >= nrows) mrow = nrows - 1;       // clamp (stores are guarded)
  const float* arow = A + (size_t)mrow * FDIM;
  const int kA = (lane < 16) ? 0 : 16;

  v16h a0, a1;
  load8h(a0, 0, arow + kA);
  load8h(a0, 8, arow + kA + 8);
  load8h(a1, 0, arow + 32 + kA);
  load8h(a1, 8, arow + 32 + kA + 8);

  // ---- B fragments: lane holds column n = lane&15, K-range of 16 ----
  const int nb = lane & 15;
  const int kB = (lane < 16) ? 0 : 16;

  v8f acc[4];
#pragma unroll
  for (int nt = 0; nt < 4; ++nt) {
    acc[nt] = (v8f){0.f, 0.f, 0.f, 0.f, 0.f, 0.f, 0.f, 0.f};
    const int n = nt * 16 + nb;
    v16h b0 = *(const v16h*)&wt[n * FDIM + kB];        // 32B aligned
    v16h b1 = *(const v16h*)&wt[n * FDIM + 32 + kB];
    acc[nt] = __builtin_amdgcn_wmma_f32_16x16x32_f16(
        false, a0, false, b0, (short)0, acc[nt], false, false);
    acc[nt] = __builtin_amdgcn_wmma_f32_16x16x32_f16(
        false, a1, false, b1, (short)0, acc[nt], false, false);
  }

  // ---- store: C/D layout — lane n = lane&15, VGPR v holds M = v + 8*(lane>=16)
  const int mBase = row0 + ((lane >> 4) << 3);
#pragma unroll
  for (int nt = 0; nt < 4; ++nt) {
    const int col = nt * 16 + nb;
#pragma unroll
    for (int v = 0; v < 8; ++v) {
      const int mi = mBase + v;
      if (mi >= nrows) continue;
      const size_t o = (size_t)mi * FDIM + col;
      float r = acc[nt][v];
      if (residual) out[o] = residual[o] + fmaxf(r, 0.f);
      else          out[o] = fmaxf(r + bias[col], 0.f);
    }
  }
}

// ---------------------------------------------------------------------------
extern "C" void kernel_launch(void* const* d_in, const int* in_sizes, int n_in,
                              void* d_out, int out_size, void* d_ws, size_t ws_size,
                              hipStream_t stream) {
  const float*     x      = (const float*)d_in[0];
  const long long* idx    = (const long long*)d_in[1];
  const float*     ew     = (const float*)d_in[2];
  /* d_in[3] = edge_attr, unused */
  const float*     lin0_w = (const float*)d_in[4];
  const float*     lin0_b = (const float*)d_in[5];
  const float*     w1     = (const float*)d_in[6];

  const long long N = in_sizes[0] / FDIM;
  const long long E = in_sizes[2];

  // workspace layout (floats)
  float* ws    = (float*)d_ws;
  float* normw = ws;                       // [E]
  float* deg   = normw + E;                // [N]
  float* h0    = deg + N;                  // [N*64]
  float* agg   = h0 + N * FDIM;            // [N*64]
  float* hbuf  = agg + N * FDIM;           // [N*64]

  const int BT = 256;
  const int eBlocks  = (int)((E + BT - 1) / BT);
  const int e16Blk   = (int)((E * 16 + BT - 1) / BT);
  const size_t n4    = (size_t)N * FDIM / 4;
  const int n4Blk    = (int)((n4 + BT - 1) / BT);
  const int ntiles   = (int)((N + 15) / 16);
  const int gBlocks  = (ntiles + 7) / 8;

  // ---- GCN normalization ----
  hipMemsetAsync(deg, 0, (size_t)N * sizeof(float), stream);
  deg_kernel<<<eBlocks, BT, 0, stream>>>(idx, ew, deg, E);
  normw_kernel<<<eBlocks, BT, 0, stream>>>(idx, ew, deg, normw, E);

  // ---- h0 = relu(x @ lin0_w^T + b) : lin0_w is [n][k] already ----
  gemm_kernel<<<gBlocks, BT, 0, stream>>>(x, lin0_w, lin0_b, nullptr,
                                          h0, (int)N, /*w_nk=*/1);

  // ---- conv layers ----
  const float* h_cur = h0;
  for (int i = 0; i < 2; ++i) {
    float* dst = (i == 0) ? hbuf : (float*)d_out;
    hipMemsetAsync(agg, 0, (size_t)N * FDIM * sizeof(float), stream);
    scatter_kernel<<<e16Blk, BT, 0, stream>>>(h_cur, normw, idx, agg, E);
    axpby_kernel<<<n4Blk, BT, 0, stream>>>(agg, h0, n4);
    // dst = h_cur + relu(agg @ w1[i]) ; w1 is [k][n]
    gemm_kernel<<<gBlocks, BT, 0, stream>>>(agg, w1 + (size_t)i * FDIM * FDIM,
                                            nullptr, h_cur, dst, (int)N,
                                            /*w_nk=*/0);
    h_cur = dst;
  }
}
